// MolecularGAT_103079215285
// MI455X (gfx1250) — matrix operand order repaired and checked by reading, not compiled, not run on hardware
//
#include <hip/hip_runtime.h>
#include <hip/hip_bf16.h>

typedef __attribute__((ext_vector_type(16))) __bf16 v16bf;
typedef __attribute__((ext_vector_type(8)))  float  v8f;

union V16U {                      // build v16bf from two 16-byte vector loads
  v16bf v;
  uint4 q[2];
};
static_assert(sizeof(V16U) == 32, "v16bf must be 32 bytes");

namespace {
constexpr int kB = 256, kN = 128, kA = 34, kE = 16, kH = 8, kD = 75;
constexpr int kHD = 600, kHDp = 608, kDp = 80;
constexpr int kBN = kB * kN;               // 32768
constexpr float kNeg = -1e9f;

// ---- workspace byte offsets -------------------------------------------------
constexpr size_t OFF_VE1   = 0;                                   // 16*8 f32
constexpr size_t OFF_VE2   = OFF_VE1 + 1024;                      // 16 f32
constexpr size_t OFF_ASRC1 = OFF_VE2 + 1024;                      // BN*8 f32
constexpr size_t OFF_ADST1 = OFF_ASRC1 + (size_t)kBN * kH * 4;
constexpr size_t OFF_ASRC2 = OFF_ADST1 + (size_t)kBN * kH * 4;    // BN f32
constexpr size_t OFF_ADST2 = OFF_ASRC2 + (size_t)kBN * 4;
constexpr size_t OFF_XP1B  = OFF_ADST2 + (size_t)kBN * 4;         // BN*608 bf16
constexpr size_t OFF_OUT1B = OFF_XP1B  + (size_t)kBN * kHDp * 2;  // BN*608 bf16
constexpr size_t OFF_XP2B  = OFF_OUT1B + (size_t)kBN * kHDp * 2;  // BN*80 bf16
constexpr size_t OFF_AE2   = OFF_XP2B  + (size_t)kBN * kDp * 2;   // B*N*N f32
} // namespace

__device__ __forceinline__ v8f wmma_bf16(v16bf a, v16bf b, v8f c) {
  return __builtin_amdgcn_wmma_f32_16x16x32_bf16(false, a, false, b,
                                                 (short)0, c, false, false);
}

// ---- K1: fold edge projection with edge attention vectors -------------------
__global__ void k_prep(const float* __restrict__ We1, const float* __restrict__ ae1,
                       const float* __restrict__ We2, const float* __restrict__ ae2,
                       float* __restrict__ ve1, float* __restrict__ ve2) {
  const int tid = threadIdx.x;
  if (tid < kE * kH) {
    const int e = tid / kH, h = tid % kH;
    float s = 0.f;
    for (int d = 0; d < kD; ++d) s += We1[e * kHD + h * kD + d] * ae1[h * kD + d];
    ve1[e * kH + h] = s;
  }
  if (tid < kE) {
    float s = 0.f;
    for (int d = 0; d < kD; ++d) s += We2[tid * kD + d] * ae2[d];
    ve2[tid] = s;
  }
}

// ---- K2: xp1 = atoms @ W1 (bf16 WMMA, K padded 34->64) ----------------------
// one wave per 16-row strip; operands staged zero-padded in LDS, uint4 builds.
__global__ void __launch_bounds__(32) k_gemm1(
    const float* __restrict__ atoms, const float* __restrict__ W1,
    const float* __restrict__ asv, const float* __restrict__ adv,
    __bf16* __restrict__ xp1b, float* __restrict__ asrc1,
    float* __restrict__ adst1) {
  __shared__ float strip[16][kHDp];                       // 38.9 KB
  __shared__ __attribute__((aligned(16))) __bf16 sAt[16][64];  // atoms, padded
  __shared__ __attribute__((aligned(16))) __bf16 sWt[16][64];  // W1 tile, n-major

  const int row0 = blockIdx.x * 16;
  const int lane = threadIdx.x;
  const int m = lane & 15, kh = lane >> 4;

  // zero-fill pads once (k >= 34 never rewritten below)
  {
    uint* z0 = (uint*)&sAt[0][0];
    uint* z1 = (uint*)&sWt[0][0];
    for (int i = lane; i < 512; i += 32) { z0[i] = 0u; z1[i] = 0u; }
  }
  // stage atoms strip: 16*34 = 544 contiguous floats, unconditional
  {
    const float* ab = atoms + (size_t)row0 * kA;
    for (int i = lane; i < 16 * kA; i += 32)
      sAt[i / kA][i % kA] = (__bf16)ab[i];
  }
  __syncthreads();

  for (int ct = 0; ct < kHDp / 16; ++ct) {
    const int cb = ct * 16;
    // stage W1 tile transposed (n-major), clamped+selected, unconditional loads
    for (int i = lane; i < kA * 16; i += 32) {
      const int k = i >> 4, n = i & 15;
      const int col = cb + n;
      const float v = W1[(size_t)k * kHD + (col < kHD ? col : kHD - 1)];
      sWt[n][k] = (__bf16)(col < kHD ? v : 0.f);
    }
    __syncthreads();

    v8f c = {0.f, 0.f, 0.f, 0.f, 0.f, 0.f, 0.f, 0.f};
    const uint4* ar = (const uint4*)&sAt[m][0];
    const uint4* br = (const uint4*)&sWt[m][0];
#pragma unroll
    for (int ks = 0; ks < 2; ++ks) {
      V16U a, bb;
      a.q[0]  = ar[4 * ks + 2 * kh];
      a.q[1]  = ar[4 * ks + 2 * kh + 1];
      bb.q[0] = br[4 * ks + 2 * kh];
      bb.q[1] = br[4 * ks + 2 * kh + 1];
      c = wmma_bf16(a.v, bb.v, c);
    }
#pragma unroll
    for (int r = 0; r < 8; ++r) strip[r + 8 * kh][cb + m] = c[r];
    __syncthreads();
  }

  // bf16 copy of xp1 (pad cols 600..607 are exact zeros from B padding)
  for (int idx = lane; idx < 16 * kHDp; idx += 32) {
    const int r = idx / kHDp, col = idx % kHDp;
    xp1b[(size_t)(row0 + r) * kHDp + col] = (__bf16)strip[r][col];
  }
  // per-row, per-head attention scalars from full-precision accumulators
  for (int task = lane; task < 16 * kH; task += 32) {
    const int r = task >> 3, h = task & 7;
    float s1 = 0.f, s2 = 0.f;
    for (int d = 0; d < kD; ++d) {
      const float v = strip[r][h * kD + d];
      s1 += v * asv[h * kD + d];
      s2 += v * adv[h * kD + d];
    }
    asrc1[(size_t)(row0 + r) * kH + h] = s1;
    adst1[(size_t)(row0 + r) * kH + h] = s2;
  }
}

// ---- K3: layer-1 logits + softmax(sources) + WMMA aggregation ---------------
// block = (batch b, 16-target tile); 8 waves = 8 heads. Also emits ae2 for K5.
__global__ void __launch_bounds__(256) k_attn1(
    const float* __restrict__ edges, const float* __restrict__ adjs,
    const float* __restrict__ ve1, const float* __restrict__ ve2,
    const float* __restrict__ asrc1, const float* __restrict__ adst1,
    const __bf16* __restrict__ xp1b, const float* __restrict__ b1,
    __bf16* __restrict__ out1b, float* __restrict__ ae2out) {
  __shared__ __attribute__((aligned(16))) __bf16 lg[kH][16][kN]; // logits->alpha
  __shared__ float sA[kN * kH];
  __shared__ float sT[16 * kH];
  __shared__ float sV1[kE * kH];
  __shared__ float sV2[kE];

  const int b   = blockIdx.x >> 3;
  const int t0  = (blockIdx.x & 7) * 16;
  const int tid = threadIdx.x;

  for (int i = tid; i < kN * kH; i += 256) sA[i] = asrc1[(size_t)b * kN * kH + i];
  if (tid < 16 * kH) sT[tid] = adst1[((size_t)b * kN + t0) * kH + tid];
  if (tid < kE * kH) sV1[tid] = ve1[tid];
  if (tid < kE)      sV2[tid] = ve2[tid];
  __syncthreads();

  // logits[s,t,h] = leaky(asrc[s,h] + adst[t,h] + edges[s,t]·ve1[:,h]), masked
  for (int pair = tid; pair < kN * 16; pair += 256) {
    const int s = pair >> 4, tt = pair & 15, t = t0 + tt;
    const size_t p = (size_t)(b * kN + s) * kN + t;
    float ev[kE];
    const float4* ep = (const float4*)(edges + p * kE);
#pragma unroll
    for (int q = 0; q < 4; ++q) {
      const float4 v = ep[q];
      ev[4 * q] = v.x; ev[4 * q + 1] = v.y; ev[4 * q + 2] = v.z; ev[4 * q + 3] = v.w;
    }
    const bool msk = adjs[p] > 0.5f;
    float a2 = 0.f;
#pragma unroll
    for (int e = 0; e < kE; ++e) a2 += ev[e] * sV2[e];
    ae2out[p] = a2;                               // layer-2 edge term (reused in K5)
#pragma unroll
    for (int h = 0; h < kH; ++h) {
      float ae = 0.f;
#pragma unroll
      for (int e = 0; e < kE; ++e) ae += ev[e] * sV1[e * kH + h];
      float l = sA[s * kH + h] + sT[tt * kH + h] + ae;
      l = (l > 0.f) ? l : 0.2f * l;
      lg[h][tt][s] = (__bf16)(msk ? l : kNeg);
    }
  }
  __syncthreads();

  // softmax over sources per (h, t); alpha written in place (bf16)
  if (tid < kH * 16) {
    const int h = tid >> 4, tt = tid & 15;
    float mx = -3.4e38f;
    for (int s = 0; s < kN; ++s) mx = fmaxf(mx, (float)lg[h][tt][s]);
    float sum = 0.f;
    for (int s = 0; s < kN; ++s) sum += __expf((float)lg[h][tt][s] - mx);
    const float inv = 1.f / sum;
    for (int s = 0; s < kN; ++s)
      lg[h][tt][s] = (__bf16)(__expf((float)lg[h][tt][s] - mx) * inv);
  }
  __syncthreads();

  // out1[t, h*75+d] = sum_s alpha[s,t,h] * xp1[s, h*75+d]   (M=t, K=s, N=d)
  const int wave = tid >> 5, lane = tid & 31;
  const int m = lane & 15, kh = lane >> 4;
  const int h = wave;
  const uint4* lr = (const uint4*)&lg[h][m][0];   // 256-B row pitch, 16-B aligned
  for (int dt = 0; dt < 5; ++dt) {
    const int d0 = dt * 16;
    const int col = h * kD + d0 + m;              // < 608 always
    v8f c = {0.f, 0.f, 0.f, 0.f, 0.f, 0.f, 0.f, 0.f};
    for (int ks = 0; ks < 4; ++ks) {
      V16U a, bm;
      a.q[0] = lr[4 * ks + 2 * kh];
      a.q[1] = lr[4 * ks + 2 * kh + 1];
#pragma unroll
      for (int j = 0; j < 16; ++j) {              // coalesces to 32-B segments
        const int s = ks * 32 + 16 * kh + j;
        bm.v[j] = xp1b[(size_t)(b * kN + s) * kHDp + col];
      }
      c = wmma_bf16(a.v, bm.v, c);
    }
    const int d = d0 + m;
    if (d < kD) {
      const int cg = h * kD + d;
      const float bias = b1[cg];
#pragma unroll
      for (int r = 0; r < 8; ++r) {
        const int tt2 = r + 8 * kh;
        out1b[(size_t)(b * kN + t0 + tt2) * kHDp + cg] = (__bf16)(c[r] + bias);
      }
    }
  }
  // deterministic zero pad (cols 600..607 feed zero-padded B rows of GEMM2)
  if (tid < 16 * 8) {
    const int r = tid >> 3, cc = kHD + (tid & 7);
    out1b[(size_t)(b * kN + t0 + r) * kHDp + cc] = (__bf16)0.f;
  }
}

// ---- K4: xp2 = out1 @ W2 (K=608), epilogue asrc2/adst2 ----------------------
__global__ void __launch_bounds__(32) k_gemm2(
    const __bf16* __restrict__ out1b, const float* __restrict__ W2,
    const float* __restrict__ as2, const float* __restrict__ ad2,
    __bf16* __restrict__ xp2b, float* __restrict__ asrc2,
    float* __restrict__ adst2) {
  __shared__ float strip[16][kDp];                              // 5 KB
  __shared__ __attribute__((aligned(16))) __bf16 sWt[16][kHDp]; // 19 KB, n-major

  const int row0 = blockIdx.x * 16;
  const int lane = threadIdx.x;
  const int m = lane & 15, kh = lane >> 4;

  {  // zero-fill once (k >= 600 never rewritten)
    uint* z = (uint*)&sWt[0][0];
    for (int i = lane; i < 16 * kHDp / 2; i += 32) z[i] = 0u;
  }
  __syncthreads();

  const uint4* arow = (const uint4*)(out1b + (size_t)(row0 + m) * kHDp);

  for (int ct = 0; ct < kDp / 16; ++ct) {
    const int cb = ct * 16;
    // stage W2 tile transposed, clamped address + select (no predicated loads)
    for (int i = lane; i < kHD * 16; i += 32) {
      const int k = i >> 4, n = i & 15;
      const int col = cb + n;
      const float v = W2[(size_t)k * kD + (col < kD ? col : kD - 1)];
      sWt[n][k] = (__bf16)(col < kD ? v : 0.f);
    }
    __syncthreads();

    const uint4* br = (const uint4*)&sWt[m][0];
    v8f c = {0.f, 0.f, 0.f, 0.f, 0.f, 0.f, 0.f, 0.f};
    for (int ks = 0; ks < kHDp / 32; ++ks) {      // 19 k-steps
      V16U a, bb;
      a.q[0]  = arow[4 * ks + 2 * kh];            // 16-B aligned global loads
      a.q[1]  = arow[4 * ks + 2 * kh + 1];
      bb.q[0] = br[4 * ks + 2 * kh];
      bb.q[1] = br[4 * ks + 2 * kh + 1];
      c = wmma_bf16(a.v, bb.v, c);
    }
#pragma unroll
    for (int r = 0; r < 8; ++r) strip[r + 8 * kh][cb + m] = c[r];
    __syncthreads();
  }

  for (int idx = lane; idx < 16 * kDp; idx += 32) {
    const int r = idx / kDp, col = idx % kDp;
    xp2b[(size_t)(row0 + r) * kDp + col] = (__bf16)strip[r][col];
  }
  if (lane < 16) {
    float s1 = 0.f, s2 = 0.f;
    for (int d = 0; d < kD; ++d) {
      const float v = strip[lane][d];
      s1 += v * as2[d];
      s2 += v * ad2[d];
    }
    asrc2[row0 + lane] = s1;
    adst2[row0 + lane] = s2;
  }
}

// ---- K5: layer-2 logits (reusing ae2) + softmax + WMMA aggregation ----------
// xp2 panel staged to LDS with CDNA5 async global->LDS b128 copies.
__global__ void __launch_bounds__(256) k_attn2(
    const float* __restrict__ adjs, const float* __restrict__ ae2,
    const float* __restrict__ asrc2, const float* __restrict__ adst2,
    const __bf16* __restrict__ xp2b, const float* __restrict__ b2,
    float* __restrict__ out) {
  __shared__ __attribute__((aligned(16))) __bf16 sXP[kN][kDp];  // 20 KB panel
  __shared__ __attribute__((aligned(16))) __bf16 lg[16][kN];
  __shared__ float sA[kN], sT[16];
  const int b = blockIdx.x >> 3, t0 = (blockIdx.x & 7) * 16, tid = threadIdx.x;

  // issue async copies of the whole per-batch xp2 panel (1280 x 16B chunks)
  {
    const __bf16* base = xp2b + (size_t)b * kN * kDp;
    const unsigned lds0 = (unsigned)(uintptr_t)&sXP[0][0];
#pragma unroll
    for (int i = 0; i < 5; ++i) {
      const int c = tid + 256 * i;                // < 1280, no row straddle
      const unsigned lds = lds0 + (unsigned)c * 16u;
      const __bf16* g = base + (size_t)c * 8;
      asm volatile("global_load_async_to_lds_b128 %0, %1, off"
                   :: "v"(lds), "v"(g) : "memory");
    }
  }

  if (tid < kN) sA[tid] = asrc2[b * kN + tid];
  if (tid < 16) sT[tid] = adst2[b * kN + t0 + tid];
  __syncthreads();

  for (int pair = tid; pair < kN * 16; pair += 256) {
    const int s = pair >> 4, tt = pair & 15, t = t0 + tt;
    const size_t p = (size_t)(b * kN + s) * kN + t;
    float l = sA[s] + sT[tt] + ae2[p];
    l = (l > 0.f) ? l : 0.2f * l;
    lg[tt][s] = (__bf16)((adjs[p] > 0.5f) ? l : kNeg);
  }
  // make async panel visible to all waves before the aggregation phase
  asm volatile("s_wait_asynccnt 0x0" ::: "memory");
  __syncthreads();

  if (tid < 16) {
    float mx = -3.4e38f;
    for (int s = 0; s < kN; ++s) mx = fmaxf(mx, (float)lg[tid][s]);
    float sum = 0.f;
    for (int s = 0; s < kN; ++s) sum += __expf((float)lg[tid][s] - mx);
    const float inv = 1.f / sum;
    for (int s = 0; s < kN; ++s)
      lg[tid][s] = (__bf16)(__expf((float)lg[tid][s] - mx) * inv);
  }
  __syncthreads();

  const int wave = tid >> 5, lane = tid & 31;
  if (wave < 5) {                                  // 5 d-tiles of 16 (D=75)
    const int m = lane & 15, kh = lane >> 4;
    const int d0 = wave * 16;
    const uint4* lr = (const uint4*)&lg[m][0];
    v8f c = {0.f, 0.f, 0.f, 0.f, 0.f, 0.f, 0.f, 0.f};
    for (int ks = 0; ks < 4; ++ks) {
      V16U a, bm;
      a.q[0] = lr[4 * ks + 2 * kh];
      a.q[1] = lr[4 * ks + 2 * kh + 1];
#pragma unroll
      for (int j = 0; j < 16; ++j) {
        const int s = ks * 32 + 16 * kh + j;
        bm.v[j] = sXP[s][d0 + m];                  // pad cols 75..79 are zeros
      }
      c = wmma_bf16(a.v, bm.v, c);
    }
    const int d = d0 + m;
    if (d < kD) {
      const float bias = b2[d];
#pragma unroll
      for (int r = 0; r < 8; ++r) {
        const int tt2 = r + 8 * kh;
        out[(size_t)(b * kN + t0 + tt2) * kD + d] = c[r] + bias;
      }
    }
  }
}

extern "C" void kernel_launch(void* const* d_in, const int* in_sizes, int n_in,
                              void* d_out, int out_size, void* d_ws, size_t ws_size,
                              hipStream_t stream) {
  (void)in_sizes; (void)n_in; (void)out_size; (void)ws_size;
  const float* atoms     = (const float*)d_in[0];
  const float* adjs      = (const float*)d_in[1];
  const float* edges     = (const float*)d_in[2];
  const float* W1        = (const float*)d_in[3];
  const float* att_src1  = (const float*)d_in[4];
  const float* att_dst1  = (const float*)d_in[5];
  const float* We1       = (const float*)d_in[6];
  const float* att_edge1 = (const float*)d_in[7];
  const float* b1        = (const float*)d_in[8];
  const float* W2        = (const float*)d_in[9];
  const float* att_src2  = (const float*)d_in[10];
  const float* att_dst2  = (const float*)d_in[11];
  const float* We2       = (const float*)d_in[12];
  const float* att_edge2 = (const float*)d_in[13];
  const float* b2        = (const float*)d_in[14];

  char* ws = (char*)d_ws;
  float*  ve1   = (float*)(ws + OFF_VE1);
  float*  ve2   = (float*)(ws + OFF_VE2);
  float*  asrc1 = (float*)(ws + OFF_ASRC1);
  float*  adst1 = (float*)(ws + OFF_ADST1);
  float*  asrc2 = (float*)(ws + OFF_ASRC2);
  float*  adst2 = (float*)(ws + OFF_ADST2);
  __bf16* xp1b  = (__bf16*)(ws + OFF_XP1B);
  __bf16* out1b = (__bf16*)(ws + OFF_OUT1B);
  __bf16* xp2b  = (__bf16*)(ws + OFF_XP2B);
  float*  ae2   = (float*)(ws + OFF_AE2);

  k_prep <<<1, 128, 0, stream>>>(We1, att_edge1, We2, att_edge2, ve1, ve2);
  k_gemm1<<<kBN / 16, 32, 0, stream>>>(atoms, W1, att_src1, att_dst1,
                                       xp1b, asrc1, adst1);
  k_attn1<<<kB * 8, 256, 0, stream>>>(edges, adjs, ve1, ve2, asrc1, adst1,
                                      xp1b, b1, out1b, ae2);
  k_gemm2<<<kBN / 16, 32, 0, stream>>>(out1b, W2, att_src2, att_dst2,
                                       xp2b, asrc2, adst2);
  k_attn2<<<kB * 8, 256, 0, stream>>>(adjs, ae2, asrc2, adst2, xp2b, b2,
                                      (float*)d_out);
}
